// AttentionLayer_69054484185528
// MI455X (gfx1250) — matrix operand order repaired and checked by reading, not compiled, not run on hardware
//
#include <hip/hip_runtime.h>

// ---------------------------------------------------------------------------
// Fused additive (Bahdanau) attention for MI455X / gfx1250 (wave32, WMMA+TDM).
// B=64, T=2048, E=D=H=1024.
//  k_prep_wt   : W_enc [E,H] fp32  ->  W_encT [H,E] bf16 (L2-resident, 2MB)
//  k_prep_u    : u[b,h] = dec@W_dec + b_dec + b_enc
//  k_scores    : scores[b,t] = tanh(enc@W_enc + u) . w_out + b_out
//                v_wmma_f32_16x16x32_bf16; S^T tiling; enc staged in LDS bf16;
//                W_encT h-chunks DMA'd to LDS via tensor_load_to_lds (TDM),
//                double-buffered with s_wait_tensorcnt; v_tanh_f32 epilogue.
//  k_softmax   : masked softmax over T -> probs (d_out)
//  k_attn_part : split-T partial weighted sums of enc (deterministic)
//  k_attn_red  : reduce partials -> attn (d_out)
// ---------------------------------------------------------------------------

typedef __attribute__((ext_vector_type(16))) __bf16       v16bf;
typedef __attribute__((ext_vector_type(4)))  __bf16       v4bf;
typedef __attribute__((ext_vector_type(8)))  float        v8f;
typedef __attribute__((ext_vector_type(4)))  unsigned int u32x4;
typedef __attribute__((ext_vector_type(8)))  unsigned int u32x8;

namespace {
constexpr int Bn = 64;
constexpr int Tn = 2048;
constexpr int En = 1024;
constexpr int Hn = 1024;
constexpr int Dn = 1024;
constexpr int LDSS  = En + 8;      // enc LDS row stride in bf16 (2064B)
constexpr int ASTR  = 2064;        // A-chunk LDS row stride in BYTES (TDM pad: 8B/1KB)
constexpr int ABUF  = 16 * ASTR;   // one A chunk buffer: 33024 B
constexpr float NEGINF = -1e9f;

union V16U { u32x4 q[2]; v16bf v; };

__device__ inline v16bf load_frag(const __bf16* p0, const __bf16* p1) {
  V16U t;
  t.q[0] = *reinterpret_cast<const u32x4*>(p0);
  t.q[1] = *reinterpret_cast<const u32x4*>(p1);
  return t.v;
}

__device__ inline unsigned short f2bf_bits(float f) {
  union { __bf16 b; unsigned short s; } c;
  c.b = (__bf16)f;
  return c.s;
}

// Native CDNA5 transcendental tanh (V_TANH_F32); v_nop covers TRANS RAW hazard.
__device__ inline float fast_tanh(float x) {
#if __has_builtin(__builtin_amdgcn_tanhf)
  return __builtin_amdgcn_tanhf(x);
#elif __has_builtin(__builtin_amdgcn_tanh_f32)
  return __builtin_amdgcn_tanh_f32(x);
#else
  float y;
  asm volatile("v_tanh_f32 %0, %1\n\tv_nop" : "=v"(y) : "v"(x));
  return y;
#endif
}

// TDM: DMA one contiguous 32KB W_encT chunk (16 h-rows x 1024 e, bf16) into LDS
// with 8B padding every 1KB (-> 2064B effective row stride, bank-spread).
// D# group0/group1 per CDNA5 ISA ch.8; 1D tile, data_size=8B, 4096 elements.
__device__ inline void tdm_issue(const unsigned short* gsrc, unsigned lds_off) {
  const unsigned long long ga = (unsigned long long)(uintptr_t)gsrc;
  u32x4 g0;
  g0[0] = 1u;                                             // count=1, user mode
  g0[1] = lds_off;                                        // lds_addr
  g0[2] = (unsigned)ga;                                   // global_addr[31:0]
  g0[3] = ((unsigned)(ga >> 32) & 0x1FFFFFFu) | (2u << 30); // addr[56:32] | type=2
  u32x8 g1;
  g1[0] = (3u << 16)        // data_size = 8B
        | (1u << 20)        // pad_enable
        | (7u << 22)        // pad_interval: 256 dwords = 1KB
        | (1u << 25);       // pad_amount: 2 dwords = 8B
  g1[1] = (4096u & 0xFFFFu) << 16;   // tensor_dim0 lo (4096 x 8B)
  g1[2] = (1u << 16);                // tensor_dim0 hi=0 | tensor_dim1=1
  g1[3] = (4096u << 16);             // tensor_dim1 hi=0 | tile_dim0=4096
  g1[4] = 1u;                        // tile_dim1=1 | tile_dim2=0
  g1[5] = 4096u;                     // tensor_dim0_stride lo
  g1[6] = 0u;
  g1[7] = 0u;
  asm volatile("tensor_load_to_lds %0, %1" :: "s"(g0), "s"(g1) : "memory");
}
} // namespace

// --- W_enc transpose + fp32->bf16 -----------------------------------------
__global__ __launch_bounds__(256) void k_prep_wt(const float* __restrict__ Wenc,
                                                 unsigned short* __restrict__ WencT) {
  __shared__ float tile[32][33];
  const int h0 = blockIdx.x * 32;
  const int e0 = blockIdx.y * 32;
  const int tx = threadIdx.x & 31;
  const int ty = threadIdx.x >> 5;
#pragma unroll
  for (int k = 0; k < 4; ++k) {
    const int e = e0 + ty + k * 8;
    tile[ty + k * 8][tx] = Wenc[(size_t)e * Hn + h0 + tx];
  }
  __syncthreads();
#pragma unroll
  for (int k = 0; k < 4; ++k) {
    const int h = h0 + ty + k * 8;
    WencT[(size_t)h * En + e0 + tx] = f2bf_bits(tile[tx][ty + k * 8]);
  }
}

// --- u[b,h] = dec[b]@W_dec + b_dec + b_enc ---------------------------------
__global__ __launch_bounds__(256) void k_prep_u(const float* __restrict__ dec,
                                                const float* __restrict__ Wdec,
                                                const float* __restrict__ benc,
                                                const float* __restrict__ bdec,
                                                float* __restrict__ u) {
  __shared__ float sdec[Dn];
  const int b = blockIdx.y;
  const int h = blockIdx.x * 256 + threadIdx.x;
  for (int i = threadIdx.x; i < Dn; i += 256) sdec[i] = dec[(size_t)b * Dn + i];
  __syncthreads();
  float acc = benc[h] + bdec[h];
#pragma unroll 4
  for (int d = 0; d < Dn; ++d) acc += sdec[d] * Wdec[(size_t)d * Hn + h];
  u[(size_t)b * Hn + h] = acc;
}

// --- fused GEMM(tanh)·w_out -> scores, WMMA bf16 + TDM ----------------------
// Block: 128 threads = 4 waves; wave w owns T-rows [t0+16w, t0+16w+16).
// S^T tiles: A = W_encT chunk in LDS (M=h, K=e), B = enc^T in LDS (K=e, N=t).
__global__ __launch_bounds__(128) void k_scores(const float* __restrict__ enc,
                                                const unsigned short* __restrict__ WencT_,
                                                const float* __restrict__ u,
                                                const float* __restrict__ Wout,
                                                const float* __restrict__ bout,
                                                float* __restrict__ scores) {
  __shared__ __attribute__((aligned(16))) __bf16 sEnc[64 * LDSS];   // 132,096 B
  __shared__ __attribute__((aligned(16))) __bf16 sA[2 * ABUF / 2];  //  66,048 B
  const int b   = blockIdx.y;
  const int t0  = blockIdx.x * 64;
  const int tid = threadIdx.x;
  const int warp = tid >> 5;
  const unsigned ldsA0 = (unsigned)(uintptr_t)&sA[0];

  // Kick off TDM for A chunk 0 while we convert enc -> bf16 into LDS.
  if (warp == 0) tdm_issue(WencT_, ldsA0);

  const float* encB = enc + ((size_t)b * Tn + t0) * En;
  for (int i = tid; i < 64 * (En / 4); i += 128) {
    const int r  = i >> 8;
    const int c4 = i & 255;
    const float4 v = reinterpret_cast<const float4*>(encB + (size_t)r * En)[c4];
    v4bf pk = { (__bf16)v.x, (__bf16)v.y, (__bf16)v.z, (__bf16)v.w };
    *reinterpret_cast<v4bf*>(&sEnc[r * LDSS + c4 * 4]) = pk;
  }

  const int lane   = tid & 31;
  const int laneHi = lane >> 4;
  const int lm     = lane & 15;
  const __bf16* bRow = &sEnc[(warp * 16 + lm) * LDSS];
  const float*  ub   = u + (size_t)b * Hn;
  const char*   sAc  = reinterpret_cast<const char*>(&sA[0]);

  float partial = 0.0f;
  for (int hc = 0; hc < Hn / 16; ++hc) {
    if (warp == 0) {
      if (hc < Hn / 16 - 1) {
        tdm_issue(WencT_ + (size_t)(hc + 1) * 16 * En,
                  ldsA0 + (unsigned)(((hc + 1) & 1) * ABUF));
        __builtin_amdgcn_s_wait_tensorcnt(1);   // chunk hc has landed
      } else {
        __builtin_amdgcn_s_wait_tensorcnt(0);
      }
    }
    __syncthreads();                            // publish chunk hc to all waves

    const char* aBuf = sAc + (hc & 1) * ABUF;
    v8f acc = {};
#pragma unroll 4
    for (int k0 = 0; k0 < En; k0 += 32) {
      const int kA  = k0 + laneHi * 8;          // A: K {0..7,16..23}/{8..15,24..31}
      const int kA2 = kA + 16;
      const int kB  = k0 + laneHi * 8;          // B: K {0..7,16..23}/{8..15,24..31}
      // TDM pad: +8B per 1KB of data -> offset = 2*k + 8*(k>>9)
      const __bf16* a0 = reinterpret_cast<const __bf16*>(
          aBuf + lm * ASTR + (kA  << 1) + ((kA  >> 9) << 3));
      const __bf16* a1 = reinterpret_cast<const __bf16*>(
          aBuf + lm * ASTR + (kA2 << 1) + ((kA2 >> 9) << 3));
      const v16bf afrag = load_frag(a0, a1);
      const v16bf bfrag = load_frag(bRow + kB, bRow + kB + 16);
      acc = __builtin_amdgcn_wmma_f32_16x16x32_bf16(false, afrag, false, bfrag,
                                                    (short)0, acc, false, false);
    }

    // acc[j] = S^T[h = hc*16 + 8*laneHi + j, t = t0 + 16*warp + lm]
    const int hbase = hc * 16 + 8 * laneHi;
    const float4 u0 = reinterpret_cast<const float4*>(ub + hbase)[0];
    const float4 u1 = reinterpret_cast<const float4*>(ub + hbase)[1];
    const float4 w0 = reinterpret_cast<const float4*>(Wout + hbase)[0];
    const float4 w1 = reinterpret_cast<const float4*>(Wout + hbase)[1];
    partial += fast_tanh(acc[0] + u0.x) * w0.x;
    partial += fast_tanh(acc[1] + u0.y) * w0.y;
    partial += fast_tanh(acc[2] + u0.z) * w0.z;
    partial += fast_tanh(acc[3] + u0.w) * w0.w;
    partial += fast_tanh(acc[4] + u1.x) * w1.x;
    partial += fast_tanh(acc[5] + u1.y) * w1.y;
    partial += fast_tanh(acc[6] + u1.z) * w1.z;
    partial += fast_tanh(acc[7] + u1.w) * w1.w;

    __syncthreads();   // all waves done reading buf[hc&1] before TDM reuses it
  }

  partial += __shfl_xor(partial, 16);           // fold the two lane-halves (h split)
  if (lane < 16) {
    scores[(size_t)b * Tn + t0 + warp * 16 + lane] = partial + bout[0];
  }
}

// --- masked softmax over T --------------------------------------------------
__global__ __launch_bounds__(256) void k_softmax(const float* __restrict__ scores,
                                                 const int* __restrict__ mask,
                                                 float* __restrict__ probs) {
  __shared__ float red[256];
  const int b = blockIdx.x;
  const int tid = threadIdx.x;
  float vals[8];
  float lmax = -3.4e38f;
#pragma unroll
  for (int i = 0; i < 8; ++i) {
    const int t = tid + i * 256;
    float s = scores[(size_t)b * Tn + t];
    if (mask[(size_t)b * Tn + t] == 0) s = NEGINF;
    vals[i] = s;
    lmax = fmaxf(lmax, s);
  }
  red[tid] = lmax;
  __syncthreads();
  for (int off = 128; off > 0; off >>= 1) {
    if (tid < off) red[tid] = fmaxf(red[tid], red[tid + off]);
    __syncthreads();
  }
  const float bmax = red[0];
  __syncthreads();
  float lsum = 0.0f;
#pragma unroll
  for (int i = 0; i < 8; ++i) { vals[i] = __expf(vals[i] - bmax); lsum += vals[i]; }
  red[tid] = lsum;
  __syncthreads();
  for (int off = 128; off > 0; off >>= 1) {
    if (tid < off) red[tid] += red[tid + off];
    __syncthreads();
  }
  const float inv = 1.0f / red[0];
#pragma unroll
  for (int i = 0; i < 8; ++i)
    probs[(size_t)b * Tn + tid + i * 256] = vals[i] * inv;
}

// --- attn partials: part[tc][b][e] = sum over 128 t of probs*enc ------------
__global__ __launch_bounds__(256) void k_attn_part(const float* __restrict__ enc,
                                                   const float* __restrict__ probs,
                                                   float* __restrict__ part) {
  const int b  = blockIdx.x;
  const int tc = blockIdx.y;
  const int e0 = threadIdx.x * 4;
  const float* encB = enc + (size_t)b * Tn * En;
  const float* pB   = probs + (size_t)b * Tn + tc * 128;
  float4 acc = make_float4(0.f, 0.f, 0.f, 0.f);
  for (int i = 0; i < 128; ++i) {
    const float p = pB[i];                       // wave-uniform
    if (p != 0.0f) {                             // skip masked rows
      const float4 v = *reinterpret_cast<const float4*>(
          encB + (size_t)(tc * 128 + i) * En + e0);
      acc.x += p * v.x; acc.y += p * v.y; acc.z += p * v.z; acc.w += p * v.w;
    }
  }
  *reinterpret_cast<float4*>(part + ((size_t)tc * Bn + b) * En + e0) = acc;
}

// --- reduce partials -> attn ------------------------------------------------
__global__ __launch_bounds__(256) void k_attn_red(const float* __restrict__ part,
                                                  float* __restrict__ attn) {
  const int idx = blockIdx.x * 256 + threadIdx.x;
  float s = 0.0f;
#pragma unroll
  for (int tc = 0; tc < 16; ++tc) s += part[(size_t)tc * Bn * En + idx];
  attn[idx] = s;
}

// ---------------------------------------------------------------------------
extern "C" void kernel_launch(void* const* d_in, const int* in_sizes, int n_in,
                              void* d_out, int out_size, void* d_ws, size_t ws_size,
                              hipStream_t stream) {
  const float* enc  = (const float*)d_in[0];
  const float* dec  = (const float*)d_in[1];
  const int*   mask = (const int*)  d_in[2];
  const float* Wenc = (const float*)d_in[3];
  const float* benc = (const float*)d_in[4];
  const float* Wdec = (const float*)d_in[5];
  const float* bdec = (const float*)d_in[6];
  const float* Wout = (const float*)d_in[7];
  const float* bout = (const float*)d_in[8];

  float* attn  = (float*)d_out;                    // [B, E]
  float* probs = (float*)d_out + (size_t)Bn * En;  // [B, T]

  char* ws = (char*)d_ws;
  unsigned short* WencT  = (unsigned short*)ws;                         // 2 MB bf16
  float*          u      = (float*)(ws + (size_t)Hn * En * 2);          // 256 KB
  float*          scores = (float*)(ws + (size_t)Hn * En * 2
                                       + (size_t)Bn * Hn * 4);          // 512 KB
  float*          part   = (float*)(ws + (size_t)Hn * En * 2
                                       + (size_t)Bn * Hn * 4
                                       + (size_t)Bn * Tn * 4);          // 4 MB

  k_prep_wt <<<dim3(Hn / 32, En / 32), 256, 0, stream>>>(Wenc, WencT);
  k_prep_u  <<<dim3(Hn / 256, Bn),     256, 0, stream>>>(dec, Wdec, benc, bdec, u);
  k_scores  <<<dim3(Tn / 64, Bn),      128, 0, stream>>>(enc, WencT, u, Wout, bout, scores);
  k_softmax <<<Bn,                     256, 0, stream>>>(scores, mask, probs);
  k_attn_part<<<dim3(Bn, Tn / 128),    256, 0, stream>>>(enc, probs, part);
  k_attn_red<<<(Bn * En) / 256,        256, 0, stream>>>(part, attn);
}
